// GATRegressor_66357244723323
// MI455X (gfx1250) — compile-verified, matrix operand order checked by reading, not run on hardware
//
#include <hip/hip_runtime.h>
#include <math.h>

typedef __bf16 bf16;
typedef __attribute__((ext_vector_type(16))) __bf16 v16bf;
typedef __attribute__((ext_vector_type(8)))  __bf16 v8bf;
typedef __attribute__((ext_vector_type(8)))  float  v8f;

// ---------------------------------------------------------------------------
// Utility kernels
// ---------------------------------------------------------------------------
__global__ void fill_f32_k(float* __restrict__ p, float v, size_t n) {
  size_t i = (size_t)blockIdx.x * blockDim.x + threadIdx.x;
  if (i < n) p[i] = v;
}

__global__ void f32_to_bf16_k(const float* __restrict__ in, bf16* __restrict__ out, size_t n) {
  size_t i = (size_t)blockIdx.x * blockDim.x + threadIdx.x;
  if (i < n) out[i] = (bf16)in[i];
}

// ---------------------------------------------------------------------------
// WMMA GEMM:  C[M,Nout] = A[M,K] * B[Nout,K]^T   (bf16 in, f32 accumulate)
// Per-wave: one 16(M) x 64(N) strip, K-loop in steps of 32, A fragment reused
// across 4 WMMAs. Requires M%16==0, K%32==0, Nout%64==0 (true here).
//
// 16-bit A/B fragment layout (CDNA5 ISA 7.12.2): lane L holds row (L&15),
// halves {K=kb..kb+7, K=kb+16..kb+23} with kb = (L<16 ? 0 : 8)  ->
// two contiguous 16-byte loads per fragment.
// ---------------------------------------------------------------------------
__device__ __forceinline__ v16bf load_frag16x32(const bf16* __restrict__ base, int stride) {
  int lane = threadIdx.x & 31;
  int r  = lane & 15;
  int kb = (lane & 16) ? 8 : 0;
  const bf16* p = base + (size_t)r * stride + kb;
  v8bf lo = *(const v8bf*)(p);        // K = kb .. kb+7
  v8bf hi = *(const v8bf*)(p + 16);   // K = kb+16 .. kb+23
  v16bf f;
#pragma unroll
  for (int i = 0; i < 8; ++i) { f[i] = lo[i]; f[i + 8] = hi[i]; }
  return f;
}

__global__ void __launch_bounds__(256)
wmma_gemm_nt_bf16_k(const bf16* __restrict__ A, const bf16* __restrict__ B,
                    float* __restrict__ C, int M, int K, int Nout) {
  int lane = threadIdx.x & 31;
  int gw = (int)(((size_t)blockIdx.x * blockDim.x + threadIdx.x) >> 5);
  int stripsN = Nout >> 6;               // 64-wide N strips
  int mTiles  = M >> 4;
  if (gw >= mTiles * stripsN) return;    // wave-uniform guard (EXEC stays all-1s)
  int mTile = gw / stripsN;
  int n0    = (gw % stripsN) << 6;

  const bf16* Abase = A + (size_t)mTile * 16 * K;
  const bf16* B0 = B + (size_t)(n0 +  0) * K;
  const bf16* B1 = B + (size_t)(n0 + 16) * K;
  const bf16* B2 = B + (size_t)(n0 + 32) * K;
  const bf16* B3 = B + (size_t)(n0 + 48) * K;

  v8f acc0 = {}, acc1 = {}, acc2 = {}, acc3 = {};
  for (int k0 = 0; k0 < K; k0 += 32) {
    v16bf a  = load_frag16x32(Abase + k0, K);
    v16bf b0 = load_frag16x32(B0 + k0, K);
    v16bf b1 = load_frag16x32(B1 + k0, K);
    v16bf b2 = load_frag16x32(B2 + k0, K);
    v16bf b3 = load_frag16x32(B3 + k0, K);
    acc0 = __builtin_amdgcn_wmma_f32_16x16x32_bf16(false, a, false, b0, (short)0, acc0, false, false);
    acc1 = __builtin_amdgcn_wmma_f32_16x16x32_bf16(false, a, false, b1, (short)0, acc1, false, false);
    acc2 = __builtin_amdgcn_wmma_f32_16x16x32_bf16(false, a, false, b2, (short)0, acc2, false, false);
    acc3 = __builtin_amdgcn_wmma_f32_16x16x32_bf16(false, a, false, b3, (short)0, acc3, false, false);
  }

  // D layout: lane holds column n=(lane&15); VGPR v holds row mb+v, mb=(lane<16?0:8)
  int n  = lane & 15;
  int mb = (lane & 16) ? 8 : 0;
  float* Cr = C + (size_t)(mTile * 16 + mb) * Nout + n0 + n;
#pragma unroll
  for (int v = 0; v < 8; ++v) {
    Cr[(size_t)v * Nout +  0] = acc0[v];
    Cr[(size_t)v * Nout + 16] = acc1[v];
    Cr[(size_t)v * Nout + 32] = acc2[v];
    Cr[(size_t)v * Nout + 48] = acc3[v];
  }
}

// ---------------------------------------------------------------------------
// Attention score dots: a_s[n,h] = dot(xh[n,h,:], att_s[h,:]), same for a_d.
// One wave per (node, head), shuffle reduction (wave32).
// ---------------------------------------------------------------------------
__global__ void attn_scores_k(const float* __restrict__ xh,
                              const float* __restrict__ att_s,
                              const float* __restrict__ att_d,
                              float* __restrict__ a_s, float* __restrict__ a_d,
                              int N, int H, int C) {
  int lane = threadIdx.x & 31;
  long long gw = (((long long)blockIdx.x * blockDim.x) + threadIdx.x) >> 5;
  if (gw >= (long long)N * H) return;
  int n = (int)(gw / H), h = (int)(gw % H);
  const float* xr = xh + ((size_t)n * H + h) * C;
  float ss = 0.f, sd = 0.f;
  for (int c = lane; c < C; c += 32) {
    float v = xr[c];
    ss += v * att_s[h * C + c];
    sd += v * att_d[h * C + c];
  }
#pragma unroll
  for (int off = 16; off > 0; off >>= 1) {
    ss += __shfl_down(ss, off, 32);
    sd += __shfl_down(sd, off, 32);
  }
  if (lane == 0) { a_s[gw] = ss; a_d[gw] = sd; }
}

// ---------------------------------------------------------------------------
// Float atomic max via integer atomics (exact for IEEE with -inf init).
// ---------------------------------------------------------------------------
__device__ __forceinline__ void atomicMaxFloat(float* addr, float value) {
  if (value >= 0.0f)
    atomicMax((int*)addr, __float_as_int(value));
  else
    atomicMin((unsigned int*)addr, __float_as_uint(value));
}

// Edges e < E come from edge_index; e in [E, E+N) are the implicit self-loops.
__global__ void edge_logits_max_k(const int* __restrict__ src, const int* __restrict__ dst,
                                  int E, int N,
                                  const float* __restrict__ a_s, const float* __restrict__ a_d,
                                  float* __restrict__ elog, float* __restrict__ mmax, int H) {
  int e = blockIdx.x * blockDim.x + threadIdx.x;
  int Etot = E + N;
  if (e >= Etot) return;
  int s = (e < E) ? src[e] : (e - E);
  int d = (e < E) ? dst[e] : (e - E);
  for (int h = 0; h < H; ++h) {
    float v = a_s[s * H + h] + a_d[d * H + h];
    v = (v > 0.f) ? v : 0.2f * v;               // leaky_relu(0.2)
    elog[(size_t)e * H + h] = v;
    atomicMaxFloat(&mmax[d * H + h], v);
  }
}

__global__ void edge_expsum_k(const int* __restrict__ src, const int* __restrict__ dst,
                              int E, int N,
                              const float* __restrict__ mmax,
                              float* __restrict__ elog, float* __restrict__ ssum, int H) {
  int e = blockIdx.x * blockDim.x + threadIdx.x;
  int Etot = E + N;
  if (e >= Etot) return;
  int d = (e < E) ? dst[e] : (e - E);
  for (int h = 0; h < H; ++h) {
    float ex = __expf(elog[(size_t)e * H + h] - mmax[d * H + h]);
    elog[(size_t)e * H + h] = ex;
    atomicAdd(&ssum[d * H + h], ex);
  }
}

// Wave per edge: out[dst] += alpha * xh[src]  (the bandwidth-dominant scatter)
__global__ void edge_aggregate_k(const int* __restrict__ src, const int* __restrict__ dst,
                                 int E, int N,
                                 const float* __restrict__ xh,
                                 const float* __restrict__ ex, const float* __restrict__ ssum,
                                 float* __restrict__ out, int H, int C) {
  int lane = threadIdx.x & 31;
  long long gw = (((long long)blockIdx.x * blockDim.x) + threadIdx.x) >> 5;
  int Etot = E + N;
  if (gw >= Etot) return;
  int e = (int)gw;
  int s = (e < E) ? src[e] : (e - E);
  int d = (e < E) ? dst[e] : (e - E);
  const float* xs = xh + (size_t)s * H * C;
  float* od = out + (size_t)d * H * C;
  for (int h = 0; h < H; ++h) {
    float alpha = ex[(size_t)e * H + h] / (ssum[d * H + h] + 1e-16f);
    for (int c = lane; c < C; c += 32)
      atomicAdd(&od[h * C + c], xs[h * C + c] * alpha);
  }
}

__global__ void bias_elu_to_bf16_k(const float* __restrict__ in, const float* __restrict__ bias,
                                   bf16* __restrict__ out, size_t n, int F) {
  size_t i = (size_t)blockIdx.x * blockDim.x + threadIdx.x;
  if (i >= n) return;
  float v = in[i] + bias[i % F];
  v = (v > 0.f) ? v : expm1f(v);
  out[i] = (bf16)v;
}

__global__ void bias_elu_f32_k(float* __restrict__ io, const float* __restrict__ bias,
                               size_t n, int F) {
  size_t i = (size_t)blockIdx.x * blockDim.x + threadIdx.x;
  if (i >= n) return;
  float v = io[i] + bias[i % F];
  io[i] = (v > 0.f) ? v : expm1f(v);
}

__global__ void pool_sum_k(const float* __restrict__ h, const int* __restrict__ batch,
                           float* __restrict__ psum, float* __restrict__ pcnt,
                           int N, int F) {
  long long i = (long long)blockIdx.x * blockDim.x + threadIdx.x;
  if (i >= (long long)N * F) return;
  int n = (int)(i / F), c = (int)(i % F);
  int g = batch[n];
  atomicAdd(&psum[(size_t)g * F + c], h[i]);
  if (c == 0) atomicAdd(&pcnt[g], 1.0f);
}

// One 128-thread block per graph: mean-pool finalize + relu(Wh1*p+bh1) + Wh2 dot
__global__ void head_mlp_k(const float* __restrict__ psum, const float* __restrict__ pcnt,
                           const float* __restrict__ Wh1, const float* __restrict__ bh1,
                           const float* __restrict__ Wh2, const float* __restrict__ bh2,
                           float* __restrict__ out, int F) {
  __shared__ float pooled[128];
  __shared__ float red[128];
  int g = blockIdx.x, t = threadIdx.x;
  float cnt = fmaxf(pcnt[g], 1.0f);
  pooled[t] = psum[(size_t)g * F + t] / cnt;
  __syncthreads();
  float acc = bh1[t];
  for (int k = 0; k < F; ++k) acc += Wh1[t * F + k] * pooled[k];
  float z = fmaxf(acc, 0.f);
  red[t] = Wh2[t] * z;
  __syncthreads();
  for (int s = 64; s > 0; s >>= 1) {
    if (t < s) red[t] += red[t + s];
    __syncthreads();
  }
  if (t == 0) out[g] = red[0] + bh2[0];
}

// ---------------------------------------------------------------------------
// Host orchestration
// ---------------------------------------------------------------------------
static inline int nblk(long long work, int t) { return (int)((work + t - 1) / t); }

extern "C" void kernel_launch(void* const* d_in, const int* in_sizes, int n_in,
                              void* d_out, int out_size, void* d_ws, size_t ws_size,
                              hipStream_t stream) {
  (void)n_in; (void)out_size; (void)ws_size;
  const int Fin = 128, Hd = 128, H1 = 4, G = 128;
  const int F1 = Hd * H1;                 // 512
  const int N = in_sizes[0] / Fin;        // 50000
  const int E = in_sizes[1] / 2;          // 800000
  const int Etot = E + N;
  const int T = 256;

  const float* x    = (const float*)d_in[0];
  const int*   srcE = (const int*)d_in[1];
  const int*   dstE = srcE + E;
  const int*   batch= (const int*)d_in[2];
  const float* W1   = (const float*)d_in[3];
  const float* as1  = (const float*)d_in[4];
  const float* ad1  = (const float*)d_in[5];
  const float* b1   = (const float*)d_in[6];
  const float* W2   = (const float*)d_in[7];
  const float* as2  = (const float*)d_in[8];
  const float* ad2  = (const float*)d_in[9];
  const float* b2   = (const float*)d_in[10];
  const float* Wh1  = (const float*)d_in[11];
  const float* bh1  = (const float*)d_in[12];
  const float* Wh2  = (const float*)d_in[13];
  const float* bh2  = (const float*)d_in[14];
  float* out = (float*)d_out;

  // ---- workspace bump allocation (256B aligned), with lifetime-based reuse
  char* ws = (char*)d_ws;
  size_t off = 0;
  auto bump = [&](size_t bytes) { size_t o = off; off += (bytes + 255) & ~(size_t)255; return o; };

  size_t poolA = bump((size_t)N * Fin * sizeof(bf16));   // xb -> layer2 smalls
  size_t poolB = bump((size_t)N * F1  * sizeof(float));  // xh1 -> h1b+xh2+agg2
  size_t oW1b  = bump((size_t)F1 * Fin * sizeof(bf16));
  size_t oW2b  = bump((size_t)Hd * F1  * sizeof(bf16));
  size_t oas1  = bump((size_t)N * H1 * sizeof(float));
  size_t oad1  = bump((size_t)N * H1 * sizeof(float));
  size_t om1   = bump((size_t)N * H1 * sizeof(float));
  size_t os1   = bump((size_t)N * H1 * sizeof(float));
  size_t oel1  = bump((size_t)Etot * H1 * sizeof(float));
  size_t oagg1 = bump((size_t)N * F1 * sizeof(float));

  bf16*  xb    = (bf16*)(ws + poolA);
  float* xh1   = (float*)(ws + poolB);
  bf16*  W1b   = (bf16*)(ws + oW1b);
  bf16*  W2b   = (bf16*)(ws + oW2b);
  float* a_s1  = (float*)(ws + oas1);
  float* a_d1  = (float*)(ws + oad1);
  float* m1    = (float*)(ws + om1);
  float* s1    = (float*)(ws + os1);
  float* elog1 = (float*)(ws + oel1);
  float* agg1  = (float*)(ws + oagg1);

  // pool B reuse (xh1 dead after layer-1 aggregation): exact fit N*2048 bytes
  bf16*  h1b  = (bf16*)(ws + poolB);
  float* xh2  = (float*)(ws + poolB + (size_t)N * F1 * sizeof(bf16));
  float* agg2 = (float*)(ws + poolB + (size_t)N * F1 * sizeof(bf16) + (size_t)N * Hd * sizeof(float));
  // pool A reuse (xb dead after GEMM1): 4.3MB < 12.8MB
  char* pa = ws + poolA;
  float* a_s2  = (float*)pa;  pa += (size_t)N * sizeof(float);
  float* a_d2  = (float*)pa;  pa += (size_t)N * sizeof(float);
  float* m2    = (float*)pa;  pa += (size_t)N * sizeof(float);
  float* s2    = (float*)pa;  pa += (size_t)N * sizeof(float);
  float* elog2 = (float*)pa;  pa += (size_t)Etot * sizeof(float);
  float* psum  = (float*)pa;  pa += (size_t)G * Hd * sizeof(float);
  float* pcnt  = (float*)pa;

  // ================= Layer 1 =================
  f32_to_bf16_k<<<nblk((long long)N * Fin, T), T, 0, stream>>>(x, xb, (size_t)N * Fin);
  f32_to_bf16_k<<<nblk((long long)F1 * Fin, T), T, 0, stream>>>(W1, W1b, (size_t)F1 * Fin);
  f32_to_bf16_k<<<nblk((long long)Hd * F1, T), T, 0, stream>>>(W2, W2b, (size_t)Hd * F1);

  { // xh1[N,512] = xb * W1b^T
    long long waves = (long long)(N / 16) * (F1 / 64);
    wmma_gemm_nt_bf16_k<<<nblk(waves * 32, T), T, 0, stream>>>(xb, W1b, xh1, N, Fin, F1);
  }

  attn_scores_k<<<nblk((long long)N * H1 * 32, T), T, 0, stream>>>(xh1, as1, ad1, a_s1, a_d1, N, H1, Hd);

  fill_f32_k<<<nblk((long long)N * H1, T), T, 0, stream>>>(m1, -INFINITY, (size_t)N * H1);
  fill_f32_k<<<nblk((long long)N * H1, T), T, 0, stream>>>(s1, 0.0f, (size_t)N * H1);
  fill_f32_k<<<nblk((long long)N * F1, T), T, 0, stream>>>(agg1, 0.0f, (size_t)N * F1);

  edge_logits_max_k<<<nblk(Etot, T), T, 0, stream>>>(srcE, dstE, E, N, a_s1, a_d1, elog1, m1, H1);
  edge_expsum_k<<<nblk(Etot, T), T, 0, stream>>>(srcE, dstE, E, N, m1, elog1, s1, H1);
  edge_aggregate_k<<<nblk((long long)Etot * 32, T), T, 0, stream>>>(srcE, dstE, E, N, xh1, elog1, s1, agg1, H1, Hd);

  bias_elu_to_bf16_k<<<nblk((long long)N * F1, T), T, 0, stream>>>(agg1, b1, h1b, (size_t)N * F1, F1);

  // ================= Layer 2 =================
  { // xh2[N,128] = h1b * W2b^T
    long long waves = (long long)(N / 16) * (Hd / 64);
    wmma_gemm_nt_bf16_k<<<nblk(waves * 32, T), T, 0, stream>>>(h1b, W2b, xh2, N, F1, Hd);
  }

  attn_scores_k<<<nblk((long long)N * 32, T), T, 0, stream>>>(xh2, as2, ad2, a_s2, a_d2, N, 1, Hd);

  fill_f32_k<<<nblk((long long)N, T), T, 0, stream>>>(m2, -INFINITY, (size_t)N);
  fill_f32_k<<<nblk((long long)N, T), T, 0, stream>>>(s2, 0.0f, (size_t)N);
  fill_f32_k<<<nblk((long long)N * Hd, T), T, 0, stream>>>(agg2, 0.0f, (size_t)N * Hd);
  fill_f32_k<<<nblk((long long)G * Hd + G, T), T, 0, stream>>>(psum, 0.0f, (size_t)G * Hd + G); // psum + pcnt (contiguous)

  edge_logits_max_k<<<nblk(Etot, T), T, 0, stream>>>(srcE, dstE, E, N, a_s2, a_d2, elog2, m2, 1);
  edge_expsum_k<<<nblk(Etot, T), T, 0, stream>>>(srcE, dstE, E, N, m2, elog2, s2, 1);
  edge_aggregate_k<<<nblk((long long)Etot * 32, T), T, 0, stream>>>(srcE, dstE, E, N, xh2, elog2, s2, agg2, 1, Hd);

  bias_elu_f32_k<<<nblk((long long)N * Hd, T), T, 0, stream>>>(agg2, b2, (size_t)N * Hd, Hd);

  // ================= Pool + head =================
  pool_sum_k<<<nblk((long long)N * Hd, T), T, 0, stream>>>(agg2, batch, psum, pcnt, N, Hd);
  head_mlp_k<<<G, Hd, 0, stream>>>(psum, pcnt, Wh1, bh1, Wh2, bh2, out, Hd);
}
